// NestedNERModule_33526514713191
// MI455X (gfx1250) — matrix operand order repaired and checked by reading, not compiled, not run on hardware
//
#include <hip/hip_runtime.h>
#include <hip/hip_bf16.h>

typedef __attribute__((ext_vector_type(16))) __bf16 v16bf;
typedef __attribute__((ext_vector_type(8)))  float  v8f;

#define IMP (-10000.0f)

// ---------------------------------------------------------------------------
// Helpers
// ---------------------------------------------------------------------------
__device__ __forceinline__ unsigned pk_bf16_trunc(float x, float y) {
  return __builtin_amdgcn_perm(__float_as_uint(y), __float_as_uint(x), 0x07060302u);
}
__device__ __forceinline__ unsigned pk_bf16_rne(float x, float y) {
  unsigned xu = __float_as_uint(x), yu = __float_as_uint(y);
  xu += 0x7FFFu + ((xu >> 16) & 1u);
  yu += 0x7FFFu + ((yu >> 16) & 1u);
  return (xu >> 16) | (yu & 0xFFFF0000u);
}
__device__ __forceinline__ float lse2(float a, float b) {
  float m = fmaxf(a, b);
  return m + __logf(__expf(a - m) + __expf(b - m));
}
__device__ __forceinline__ float lse3(float a, float b, float c) {
  float m = fmaxf(a, fmaxf(b, c));
  return m + __logf(__expf(a - m) + __expf(b - m) + __expf(c - m));
}

// ---------------------------------------------------------------------------
// Kernel 0: pre-pack W into the per-lane bf16 B-fragment layout (128 KB).
// ---------------------------------------------------------------------------
__global__ __launch_bounds__(256) void pack_w_kernel(
    const float* __restrict__ W, unsigned* __restrict__ Wpk) {
  const int tid  = blockIdx.x * 256 + threadIdx.x;   // 0..32767
  const int r    = tid & 7;
  const int lane = (tid >> 3) & 31;
  const int nt   = (tid >> 8) & 7;
  const int kc   = tid >> 11;
  const int wrow = nt * 16 + (lane & 15);
  const int kb   = kc * 32 + ((r & 3) << 1) + ((r >> 2) << 4) + ((lane >> 4) << 3);
  unsigned v = 0u;
  if (wrow < 120) {
    const float2 f = *(const float2*)(W + (size_t)wrow * 512 + kb);
    v = pk_bf16_rne(f.x, f.y);
  }
  Wpk[tid] = v;
}

// ---------------------------------------------------------------------------
// Kernel 1: logits GEMM, v_wmma_f32_16x16x32_bf16.
// W fragments staged through LDS (two 64 KB phases); B fragments explicitly
// double-buffered so the ds_load for nt+1 issues before the WMMA for nt.
// ---------------------------------------------------------------------------
__global__ __launch_bounds__(256) void gemm_logits_kernel(
    const float* __restrict__ E, const unsigned* __restrict__ Wpk,
    const float* __restrict__ bias, float* __restrict__ logits) {
  const int lane = threadIdx.x & 31;
  const int wave = threadIdx.x >> 5;
  const int m0   = (blockIdx.x * 8 + wave) * 16;
  const int lm   = lane & 15;
  const int kh   = lane >> 4;

  union Frag { v16bf v; unsigned u[8]; uint4 q[2]; };

  __shared__ uint4 lds4[4096];  // 64 KB: half of Wpk per phase

  v8f acc[8];
#pragma unroll
  for (int nt = 0; nt < 8; ++nt)
#pragma unroll
    for (int i = 0; i < 8; ++i) acc[nt][i] = 0.0f;

  const float* erow = E + (size_t)(m0 + lm) * 512 + (kh << 3);

  for (int ph = 0; ph < 2; ++ph) {
    // ---- stage 64 KB of Wpk into LDS (coalesced b128 -> ds_store_b128)
    {
      const uint4* src = (const uint4*)(Wpk + ph * 16384);
#pragma unroll
      for (int i = 0; i < 4; ++i)
        lds4[i * 256 + threadIdx.x] = src[i * 256 + threadIdx.x];
    }
    __syncthreads();

#pragma unroll
    for (int kcl = 0; kcl < 8; ++kcl) {
      const int k0 = ((ph * 8 + kcl) << 5);
      // ---- A fragment: 4x b128 global loads + 8x v_perm packs
      const float4 f0 = *(const float4*)(erow + k0);
      const float4 f1 = *(const float4*)(erow + k0 + 4);
      const float4 f2 = *(const float4*)(erow + k0 + 16);
      const float4 f3 = *(const float4*)(erow + k0 + 20);
      Frag A;
      A.u[0] = pk_bf16_trunc(f0.x, f0.y);
      A.u[1] = pk_bf16_trunc(f0.z, f0.w);
      A.u[2] = pk_bf16_trunc(f1.x, f1.y);
      A.u[3] = pk_bf16_trunc(f1.z, f1.w);
      A.u[4] = pk_bf16_trunc(f2.x, f2.y);
      A.u[5] = pk_bf16_trunc(f2.z, f2.w);
      A.u[6] = pk_bf16_trunc(f3.x, f3.y);
      A.u[7] = pk_bf16_trunc(f3.z, f3.w);

      // ---- B fragments from LDS, explicit ping-pong (load nt+1 before wmma nt)
      const uint4* bbase = &lds4[(size_t)(kcl * 8) * 64 + lane * 2];
      Frag BA, BB;
      BA.q[0] = bbase[0];
      BA.q[1] = bbase[1];
#pragma unroll
      for (int nt = 0; nt < 8; ++nt) {
        if ((nt & 1) == 0) {
          if (nt < 7) { BB.q[0] = bbase[(nt + 1) * 64]; BB.q[1] = bbase[(nt + 1) * 64 + 1]; }
          acc[nt] = __builtin_amdgcn_wmma_f32_16x16x32_bf16(
              false, A.v, false, BA.v, (short)0, acc[nt], false, false);
        } else {
          if (nt < 7) { BA.q[0] = bbase[(nt + 1) * 64]; BA.q[1] = bbase[(nt + 1) * 64 + 1]; }
          acc[nt] = __builtin_amdgcn_wmma_f32_16x16x32_bf16(
              false, A.v, false, BB.v, (short)0, acc[nt], false, false);
        }
      }
    }
    __syncthreads();
  }

#pragma unroll
  for (int nt = 0; nt < 8; ++nt) {
    const int col = nt * 16 + lm;
    if (col < 120) {
      const float bc = bias[col];
#pragma unroll
      for (int r = 0; r < 8; ++r) {
        const int row = m0 + r + kh * 8;
        logits[(size_t)row * 120 + col] = acc[nt][r] + bc;
      }
    }
  }
}

// ---------------------------------------------------------------------------
// CRF scan machinery: software-pipelined 8-step blocks with ping-pong
// register buffers so the (recurrence-independent) logit/mask loads for the
// next block overlap the exp/log chain of the current block.
// ---------------------------------------------------------------------------
struct FSt { float a0, a1, a2, a3, a4; };
struct BSt { float b0, b1, b2, b3, b4; };

__device__ __forceinline__ void load_e8(const float* e, int tbase, float (&be)[8][5]) {
#pragma unroll
  for (int i = 0; i < 8; ++i) {
    const float* p = e + (size_t)(tbase + i) * 120;
    be[i][0] = p[0]; be[i][1] = p[1]; be[i][2] = p[2]; be[i][3] = p[3]; be[i][4] = p[4];
  }
}
__device__ __forceinline__ void load_m8(const unsigned char* mk, int tbase,
                                        unsigned char (&bm)[8]) {
#pragma unroll
  for (int i = 0; i < 8; ++i) bm[i] = mk[tbase + i];
}

__device__ __forceinline__ void fwd_step(FSt& s, const float* et, bool mt, float* o) {
  const float F3 = lse3(s.a0, s.a3, s.a4);  // to O/B/U from {O,L,U}
  const float F2 = lse2(s.a1, s.a2);        // to I/L   from {I,B}
  s.a0 = mt ? F3 + et[0] : s.a0;
  s.a1 = mt ? F2 + et[1] : s.a1;
  s.a2 = mt ? F3 + et[2] : s.a2;
  s.a3 = mt ? F2 + et[3] : s.a3;
  s.a4 = mt ? F3 + et[4] : s.a4;
  o[0] = s.a0; o[1] = s.a1; o[2] = s.a2; o[3] = s.a3; o[4] = s.a4;
}
__device__ __forceinline__ void fwd_blk8(FSt& s, const float (&be)[8][5],
                                         const unsigned char (&bm)[8],
                                         float* out, int tbase) {
#pragma unroll
  for (int i = 0; i < 8; ++i)
    fwd_step(s, be[i], bm[i] != 0, out + (size_t)(tbase + i) * 5);
}

__device__ __forceinline__ void bwd_step(BSt& s, const float* en, bool mt, bool mn,
                                         float* o) {
  const float x0 = en[0] + s.b0, x1 = en[1] + s.b1, x2 = en[2] + s.b2;
  const float x3 = en[3] + s.b3, x4 = en[4] + s.b4;
  const float A3 = lse3(x0, x2, x4);  // from O/L/U: succ {O,B,U}
  const float A2 = lse2(x1, x3);      // from I/B:   succ {I,L}
  const bool r = mt && !mn;
  s.b0 = r ? 0.0f : A3;
  s.b1 = r ? IMP  : A2;
  s.b2 = r ? IMP  : A2;
  s.b3 = r ? 0.0f : A3;
  s.b4 = r ? 0.0f : A3;
  o[0] = s.b0; o[1] = s.b1; o[2] = s.b2; o[3] = s.b3; o[4] = s.b4;
}
// rows be[i] = e[tbase+i]; steps t = tbase+6 .. tbase-1 (descending, i = 7..0)
__device__ __forceinline__ void bwd_blk8(BSt& s, const float (&be)[8][5],
                                         const unsigned char (&bm)[8], unsigned char mx,
                                         float* out, int tbase) {
#pragma unroll
  for (int i = 7; i >= 0; --i) {
    const bool mt = (i > 0) ? (bm[i - 1] != 0) : (mx != 0);  // mask[tbase+i-1]
    const bool mn = bm[i] != 0;                              // mask[tbase+i]
    bwd_step(s, be[i], mt, mn, out + (size_t)(tbase + i - 1) * 5);
  }
}

// ---------------------------------------------------------------------------
// Kernel 2: fused CRF forward + backward scans (tid<1536: bwd, else fwd).
// ---------------------------------------------------------------------------
__global__ __launch_bounds__(256) void crf_scan_kernel(
    const float* __restrict__ logits, const unsigned char* __restrict__ mask,
    float* __restrict__ alphas, float* __restrict__ betas) {
  const int tid = blockIdx.x * 256 + threadIdx.x;
  const bool fwd = tid >= 1536;
  const int seq = fwd ? tid - 1536 : tid;
  const int n = seq / 24, lab = seq % 24;
  const float* e = logits + (size_t)n * 1024 * 120 + lab * 5;
  const unsigned char* mk = mask + (size_t)n * 1024;

  float beA[8][5], beB[8][5];
  unsigned char bmA[8], bmB[8];

  if (fwd) {
    float* out = alphas + (size_t)seq * 1024 * 5;
    FSt s = {e[0], IMP + e[1], e[2], IMP + e[3], e[4]};  // start + e0
    out[0] = s.a0; out[1] = s.a1; out[2] = s.a2; out[3] = s.a3; out[4] = s.a4;
    // blocks b=0..126: tbase = 1+8b covers t = 1..1016; remainder t = 1017..1023
    load_e8(e, 1, beA); load_m8(mk, 1, bmA);
    for (int b = 0; b < 127; b += 2) {
      int tb1 = 1 + 8 * (b + 1); if (tb1 > 1009) tb1 = 1009;
      load_e8(e, tb1, beB); load_m8(mk, tb1, bmB);
      fwd_blk8(s, beA, bmA, out, 1 + 8 * b);
      int tb2 = 1 + 8 * (b + 2); if (tb2 > 1009) tb2 = 1009;
      load_e8(e, tb2, beA); load_m8(mk, tb2, bmA);
      if (b + 1 < 127) fwd_blk8(s, beB, bmB, out, 1 + 8 * (b + 1));
    }
    for (int t = 1017; t < 1024; ++t)
      fwd_step(s, e + (size_t)t * 120, mk[t] != 0, out + (size_t)t * 5);
  } else {
    float* out = betas + (size_t)seq * 1024 * 5;
    BSt s = {0.0f, IMP, IMP, 0.0f, 0.0f};  // end
    // peel t = 1023 (en = 0, mn = false)
    {
      const float A3 = lse3(s.b0, s.b2, s.b4);
      const float A2 = lse2(s.b1, s.b3);
      const bool r = mk[1023] != 0;
      s.b0 = r ? 0.0f : A3; s.b1 = r ? IMP : A2; s.b2 = r ? IMP : A2;
      s.b3 = r ? 0.0f : A3; s.b4 = r ? 0.0f : A3;
      float* o = out + (size_t)1023 * 5;
      o[0] = s.b0; o[1] = s.b1; o[2] = s.b2; o[3] = s.b3; o[4] = s.b4;
    }
    // blocks b=0..126: tbase = 1016-8b; steps t = tbase+6 .. tbase-1 (1022..7)
    unsigned char mxA, mxB;
    load_e8(e, 1016, beA); load_m8(mk, 1016, bmA); mxA = mk[1015];
    for (int b = 0; b < 127; b += 2) {
      int tb1 = 1016 - 8 * (b + 1); if (tb1 < 1) tb1 = 1;
      load_e8(e, tb1, beB); load_m8(mk, tb1, bmB); mxB = mk[tb1 - 1];
      bwd_blk8(s, beA, bmA, mxA, out, 1016 - 8 * b);
      int tb2 = 1016 - 8 * (b + 2); if (tb2 < 1) tb2 = 1;
      load_e8(e, tb2, beA); load_m8(mk, tb2, bmA); mxA = mk[tb2 - 1];
      if (b + 1 < 127) bwd_blk8(s, beB, bmB, mxB, out, tb1);
    }
    // remainder t = 6..0 (en = e[t+1])
    for (int t = 6; t >= 0; --t)
      bwd_step(s, e + (size_t)(t + 1) * 120, mk[t] != 0, mk[t + 1] != 0,
               out + (size_t)t * 5);
  }
}

// ---------------------------------------------------------------------------
// Kernel 3: loss = -sum over masked (seq,t) of log_softmax(alpha+beta)[tag] / 100
// ---------------------------------------------------------------------------
__global__ __launch_bounds__(256) void crf_loss_kernel(
    const float* __restrict__ alphas, const float* __restrict__ betas,
    const int* __restrict__ tags, const unsigned char* __restrict__ mask,
    float* __restrict__ out) {
  const int g = blockIdx.x * 256 + threadIdx.x;
  const int seq = g >> 10, t = g & 1023;
  const int n = seq / 24;
  float val = 0.0f;
  if (mask[(size_t)n * 1024 + t]) {
    const float* ap = alphas + (size_t)g * 5;
    const float* bp = betas  + (size_t)g * 5;
    const float s0 = ap[0] + bp[0], s1 = ap[1] + bp[1], s2 = ap[2] + bp[2];
    const float s3 = ap[3] + bp[3], s4 = ap[4] + bp[4];
    const float m = fmaxf(fmaxf(fmaxf(s0, s1), fmaxf(s2, s3)), s4);
    const float l = m + __logf(__expf(s0 - m) + __expf(s1 - m) + __expf(s2 - m) +
                               __expf(s3 - m) + __expf(s4 - m));
    const int tg = tags[g];
    const float st = (tg == 0) ? s0 : (tg == 1) ? s1 : (tg == 2) ? s2
                   : (tg == 3) ? s3 : s4;
    val = st - l;
  }
  __shared__ float red[256];
  red[threadIdx.x] = val;
  __syncthreads();
#pragma unroll
  for (int s = 128; s > 0; s >>= 1) {
    if (threadIdx.x < s) red[threadIdx.x] += red[threadIdx.x + s];
    __syncthreads();
  }
  if (threadIdx.x == 0) atomicAdd(out, red[0] * (-0.01f));
}

// ---------------------------------------------------------------------------
extern "C" void kernel_launch(void* const* d_in, const int* in_sizes, int n_in,
                              void* d_out, int out_size, void* d_ws, size_t ws_size,
                              hipStream_t stream) {
  (void)in_sizes; (void)n_in; (void)out_size; (void)ws_size;
  const float* embeds        = (const float*)d_in[0];
  const float* W             = (const float*)d_in[1];
  const float* bias          = (const float*)d_in[2];
  const unsigned char* mask  = (const unsigned char*)d_in[3];
  const int* tags            = (const int*)d_in[4];

  float* logits = (float*)d_ws;
  float* alphas = logits + (size_t)65536 * 120;
  float* betas  = alphas + (size_t)1536 * 1024 * 5;
  unsigned* Wpk = (unsigned*)(betas + (size_t)1536 * 1024 * 5);

  hipMemsetAsync(d_out, 0, sizeof(float), stream);
  pack_w_kernel<<<128, 256, 0, stream>>>(W, Wpk);
  gemm_logits_kernel<<<512, 256, 0, stream>>>(embeds, Wpk, bias, logits);
  crf_scan_kernel<<<12, 256, 0, stream>>>(logits, mask, alphas, betas);
  crf_loss_kernel<<<6144, 256, 0, stream>>>(alphas, betas, tags, mask, (float*)d_out);
}
